// PointTransformerAutoencoder_21912923144352
// MI455X (gfx1250) — matrix operand order repaired and checked by reading, not compiled, not run on hardware
//
#include <hip/hip_runtime.h>
#include <hip/hip_bf16.h>

typedef __attribute__((ext_vector_type(16))) _Float16 v16h;
typedef __attribute__((ext_vector_type(8)))  float    v8f;

#define BB 4
#define SS 8
#define NN 1024
#define DD 3
#define HH 64
#define KK 16
#define LAT 32
#define P_TOTAL (BB*SS*NN)   // 32768 rows

// ---------------------------------------------------------------------------
// Shared helpers for the 64x64 WMMA GEMMs.
//
// A-frag (16x32 f16), CDNA5 layout: lane<16 holds K {0..7,16..23} of row
// (lane&15); lane>=16 holds K {8..15,24..31}. Loaded as 4x b128 per fragment.
// ---------------------------------------------------------------------------
struct alignas(16) F4 { float v[4]; };

__device__ __forceinline__ void load_afrags(const float* __restrict__ xr, int khi,
                                            v16h& a0, v16h& a1) {
    const F4* x4 = reinterpret_cast<const F4*>(xr);
    const int b = khi >> 2;                       // khi in {0,8} -> b in {0,2}
    F4 g0 = x4[b],     g1 = x4[b + 1], g2 = x4[b + 4],  g3 = x4[b + 5];
    F4 h0 = x4[b + 8], h1 = x4[b + 9], h2 = x4[b + 12], h3 = x4[b + 13];
#pragma unroll
    for (int i = 0; i < 4; ++i) {
        a0[i]      = (_Float16)g0.v[i];
        a0[4 + i]  = (_Float16)g1.v[i];
        a0[8 + i]  = (_Float16)g2.v[i];
        a0[12 + i] = (_Float16)g3.v[i];
        a1[i]      = (_Float16)h0.v[i];
        a1[4 + i]  = (_Float16)h1.v[i];
        a1[8 + i]  = (_Float16)h2.v[i];
        a1[12 + i] = (_Float16)h3.v[i];
    }
}

// Swizzled LDS image of W[64x64]: for (col n, half) the 32 K-halves a lane
// needs are contiguous: swz[((n*2+half)*32) + t],
//   u = t&15: kl = u<8 ? half*8 + u : 16 + half*8 + (u-8)
//   t<16  -> K = kl (k-tile 0); t>=16 -> K = 32 + kl (k-tile 1)
__device__ __forceinline__ void fill_wswz(const float* __restrict__ W,
                                          _Float16* __restrict__ swz,
                                          int tid, int nthreads) {
    for (int i = tid; i < 4096; i += nthreads) {
        int n = i >> 6, half = (i >> 5) & 1, t = i & 31;
        int kb = half << 3, u = t & 15;
        int K = ((t >> 4) << 5) + ((u < 8) ? (kb + u) : (16 + kb + (u - 8)));
        swz[i] = (_Float16)W[K * 64 + n];
    }
}

// ---------------------------------------------------------------------------
// WMMA GEMM: Y[M x 64] = act(X[M x 64] @ W[64 x 64] + b) (+ residual R)
// One wave computes a 16x64 row strip: 4 n-tiles, K=64 via two WMMAs each.
// RELU/RES are compile-time -> straight-line epilogue.
// ---------------------------------------------------------------------------
template <bool RELU, bool RES>
__global__ void gemm64_wmma(const float* __restrict__ X, const float* __restrict__ W,
                            const float* __restrict__ bias, const float* __restrict__ R,
                            float* __restrict__ Y) {
    __shared__ _Float16 sW[4096];                 // swizzled W (8 KB)
    fill_wswz(W, sW, threadIdx.x, blockDim.x);
    __syncthreads();

    const int wave = threadIdx.x >> 5;
    const int lane = threadIdx.x & 31;
    const int mtile = blockIdx.x * 8 + wave;
    const int half = lane >> 4;
    const size_t row = (size_t)mtile * 16 + (lane & 15);

    v16h a0, a1;
    load_afrags(X + row * 64, half * 8, a0, a1);

#pragma unroll
    for (int nt = 0; nt < 4; ++nt) {
        const int n = nt * 16 + (lane & 15);
        const v16h* pb = (const v16h*)(sW + (((n << 1) | half) << 5));
        v16h b0 = pb[0], b1 = pb[1];
        v8f c = {};
        c = __builtin_amdgcn_wmma_f32_16x16x32_f16(false, a0, false, b0, (short)0, c, false, false);
        c = __builtin_amdgcn_wmma_f32_16x16x32_f16(false, a1, false, b1, (short)0, c, false, false);

        const int rbase = mtile * 16 + (half << 3);
        const float bn = bias[n];
#pragma unroll
        for (int r = 0; r < 8; ++r) {
            size_t rr = (size_t)(rbase + r);
            float v = c[r] + bn;
            if (RELU) v = fmaxf(v, 0.0f);
            if (RES) v += R[rr * 64 + n];
            Y[rr * 64 + n] = v;
        }
    }
}

// ---------------------------------------------------------------------------
// Fused q/k/v WMMA GEMM: reuses A-fragments across three weight matrices
// (24 WMMAs per wave; 3x arithmetic intensity on the streamed activations).
// ---------------------------------------------------------------------------
__global__ void gemm64_qkv_wmma(const float* __restrict__ X,
                                const float* __restrict__ Wq, const float* __restrict__ bq,
                                const float* __restrict__ Wk, const float* __restrict__ bk,
                                const float* __restrict__ Wv, const float* __restrict__ bv,
                                float* __restrict__ Q, float* __restrict__ Kb,
                                float* __restrict__ V) {
    __shared__ _Float16 sW[3 * 4096];             // 24 KB swizzled weights
    fill_wswz(Wq, sW,        threadIdx.x, blockDim.x);
    fill_wswz(Wk, sW + 4096, threadIdx.x, blockDim.x);
    fill_wswz(Wv, sW + 8192, threadIdx.x, blockDim.x);
    __syncthreads();

    const int wave = threadIdx.x >> 5;
    const int lane = threadIdx.x & 31;
    const int mtile = blockIdx.x * 8 + wave;
    const int half = lane >> 4;
    const size_t row = (size_t)mtile * 16 + (lane & 15);

    v16h a0, a1;
    load_afrags(X + row * 64, half * 8, a0, a1);

    const float* biases[3] = { bq, bk, bv };
    float* outs[3] = { Q, Kb, V };
#pragma unroll
    for (int m = 0; m < 3; ++m) {
        const _Float16* wbase = sW + m * 4096;
        const float* bb = biases[m];
        float* Y = outs[m];
#pragma unroll
        for (int nt = 0; nt < 4; ++nt) {
            const int n = nt * 16 + (lane & 15);
            const v16h* pb = (const v16h*)(wbase + (((n << 1) | half) << 5));
            v16h b0 = pb[0], b1 = pb[1];
            v8f c = {};
            c = __builtin_amdgcn_wmma_f32_16x16x32_f16(false, a0, false, b0, (short)0, c, false, false);
            c = __builtin_amdgcn_wmma_f32_16x16x32_f16(false, a1, false, b1, (short)0, c, false, false);
            const int rbase = mtile * 16 + (half << 3);
            const float bn = bb[n];
#pragma unroll
            for (int r = 0; r < 8; ++r)
                Y[(size_t)(rbase + r) * 64 + n] = c[r] + bn;
        }
    }
}

// ---------------------------------------------------------------------------
// dense 3 -> 64 (enc_in and positional encodings). thread per (point, channel)
// ---------------------------------------------------------------------------
__global__ void dense3_kernel(const float* __restrict__ pts, const float* __restrict__ W,
                              const float* __restrict__ b, float* __restrict__ out) {
    int i = blockIdx.x * blockDim.x + threadIdx.x;      // p*64 + h
    int h = i & 63;
    int p = i >> 6;
    const float* pp = pts + (size_t)p * 3;
    out[i] = pp[0] * W[h] + pp[1] * W[64 + h] + pp[2] * W[128 + h] + b[h];
}

// ---------------------------------------------------------------------------
// kNN: per point, 16 nearest (squared dist) among its 1024-point cloud.
// ---------------------------------------------------------------------------
__global__ void knn_kernel(const float* __restrict__ pts, int* __restrict__ idx) {
    __shared__ float spos[NN * 3];                      // 12 KB
    const int bs = blockIdx.x >> 2;
    const int chunk = blockIdx.x & 3;
    const float* base = pts + (size_t)bs * NN * 3;
    for (int i = threadIdx.x; i < NN * 3; i += blockDim.x) spos[i] = base[i];
    __syncthreads();

    const int p = chunk * 256 + threadIdx.x;
    const float px = spos[p * 3 + 0], py = spos[p * 3 + 1], pz = spos[p * 3 + 2];

    float bd[KK]; int bi[KK];
#pragma unroll
    for (int t = 0; t < KK; ++t) { bd[t] = 3.4e38f; bi[t] = 0; }

    for (int j = 0; j < NN; ++j) {
        float dx = spos[j * 3 + 0] - px;
        float dy = spos[j * 3 + 1] - py;
        float dz = spos[j * 3 + 2] - pz;
        float d = dx * dx + dy * dy + dz * dz;
        if (d < bd[KK - 1]) {
            float nd = d; int ni = j;
#pragma unroll
            for (int t = 0; t < KK; ++t) {
                bool lt = nd < bd[t];
                float td = bd[t]; int ti = bi[t];
                if (lt) { bd[t] = nd; bi[t] = ni; nd = td; ni = ti; }
            }
        }
    }
    int gp = bs * NN + p;
#pragma unroll
    for (int t = 0; t < KK; ++t) idx[(size_t)gp * KK + t] = bi[t];
}

// ---------------------------------------------------------------------------
// Point-transformer neighborhood attention. One wave32 per point.
// ---------------------------------------------------------------------------
__global__ void ptattn_kernel(const float* __restrict__ q, const float* __restrict__ kbuf,
                              const float* __restrict__ vbuf, const float* __restrict__ pe,
                              const float* __restrict__ pts, const int* __restrict__ idx,
                              const float* __restrict__ Wpde, const float* __restrict__ bpde,
                              const float* __restrict__ Wattn, const float* __restrict__ battn,
                              float* __restrict__ out) {
    const int wave = threadIdx.x >> 5;
    const int lane = threadIdx.x & 31;
    const int p = blockIdx.x * 8 + wave;
    const int bs = p >> 10;
    const int h0 = lane, h1 = lane + 32;

    const float qpe0 = q[(size_t)p * 64 + h0] + pe[(size_t)p * 64 + h0];
    const float qpe1 = q[(size_t)p * 64 + h1] + pe[(size_t)p * 64 + h1];
    const float px = pts[p * 3 + 0], py = pts[p * 3 + 1], pz = pts[p * 3 + 2];
    const float wa0 = Wattn[h0], wa1 = Wattn[h1];
    const float bb0 = battn[0];

    float logit[KK]; int nb[KK];
#pragma unroll
    for (int n = 0; n < KK; ++n) {
        int j = idx[(size_t)p * KK + n];
        int g = (bs << 10) + j;
        nb[n] = g;
        float dx = pts[g * 3 + 0] - px;
        float dy = pts[g * 3 + 1] - py;
        float dz = pts[g * 3 + 2] - pz;
        float pde0 = dx * Wpde[h0] + dy * Wpde[64 + h0] + dz * Wpde[128 + h0] + bpde[h0];
        float pde1 = dx * Wpde[h1] + dy * Wpde[64 + h1] + dz * Wpde[128 + h1] + bpde[h1];
        float t0 = kbuf[(size_t)g * 64 + h0] + pe[(size_t)g * 64 + h0] + pde0;
        float t1 = kbuf[(size_t)g * 64 + h1] + pe[(size_t)g * 64 + h1] + pde1;
        float c = qpe0 * t0 * wa0 + qpe1 * t1 * wa1;
#pragma unroll
        for (int s = 16; s > 0; s >>= 1) c += __shfl_xor(c, s, 32);
        logit[n] = c + bb0;
    }
    float mx = logit[0];
#pragma unroll
    for (int n = 1; n < KK; ++n) mx = fmaxf(mx, logit[n]);
    float w[KK]; float se = 0.0f;
#pragma unroll
    for (int n = 0; n < KK; ++n) { w[n] = expf(logit[n] - mx); se += w[n]; }
    float inv = 1.0f / se;
    float o0 = 0.0f, o1 = 0.0f;
#pragma unroll
    for (int n = 0; n < KK; ++n) {
        float a = w[n] * inv;
        o0 += a * vbuf[(size_t)nb[n] * 64 + h0];
        o1 += a * vbuf[(size_t)nb[n] * 64 + h1];
    }
    out[(size_t)p * 64 + h0] = o0;
    out[(size_t)p * 64 + h1] = o1;
}

// ---------------------------------------------------------------------------
// LayerNorm over 64 channels, wave32 per point, in place.
// ---------------------------------------------------------------------------
__global__ void ln_kernel(float* __restrict__ x, const float* __restrict__ sc,
                          const float* __restrict__ bi) {
    const int wave = threadIdx.x >> 5;
    const int lane = threadIdx.x & 31;
    const int p = blockIdx.x * 8 + wave;
    float v0 = x[(size_t)p * 64 + lane];
    float v1 = x[(size_t)p * 64 + lane + 32];
    float s = v0 + v1;
#pragma unroll
    for (int i = 16; i > 0; i >>= 1) s += __shfl_xor(s, i, 32);
    float m = s * (1.0f / 64.0f);
    float d0 = v0 - m, d1 = v1 - m;
    float vv = d0 * d0 + d1 * d1;
#pragma unroll
    for (int i = 16; i > 0; i >>= 1) vv += __shfl_xor(vv, i, 32);
    float inv = 1.0f / sqrtf(vv * (1.0f / 64.0f) + 1e-6f);
    x[(size_t)p * 64 + lane]      = d0 * inv * sc[lane]      + bi[lane];
    x[(size_t)p * 64 + lane + 32] = d1 * inv * sc[lane + 32] + bi[lane + 32];
}

// max over N points -> xs[B*S][64]
__global__ void maxpool_n_kernel(const float* __restrict__ x, float* __restrict__ xs) {
    int i = blockIdx.x * blockDim.x + threadIdx.x;  // bs*64 + h
    int h = i & 63;
    int bs = i >> 6;
    float m = -3.4e38f;
    const float* base = x + (size_t)bs * NN * 64 + h;
    for (int n = 0; n < NN; ++n) m = fmaxf(m, base[(size_t)n * 64]);
    xs[i] = m;
}

// ---------------------------------------------------------------------------
// etl token attention (S=8) + LN + maxpool over S + latent + dec_in.
// ---------------------------------------------------------------------------
__global__ void etl_kernel(const float* __restrict__ xs,
                           const float* Wq, const float* bq,
                           const float* Wk, const float* bk,
                           const float* Wv, const float* bv,
                           const float* Wo1, const float* bo1,
                           const float* Wo2, const float* bo2,
                           const float* lns, const float* lnb,
                           const float* Wlat, const float* blat,
                           const float* Wdin, const float* bdin,
                           float* __restrict__ decx) {
    __shared__ float xb[SS][64], qb[SS][64], kb[SS][64], vb[SS][64];
    __shared__ float ob[SS][64], hb[SS][64], aw[SS][SS], mvar[SS][2], xm[64], latb[LAT];
    const int b = blockIdx.x;
    const int h = threadIdx.x;

    for (int t = 0; t < SS; ++t) xb[t][h] = xs[((size_t)b * SS + t) * 64 + h];
    __syncthreads();
    for (int t = 0; t < SS; ++t) {
        float aq = bq[h], ak = bk[h], av = bv[h];
        for (int g = 0; g < 64; ++g) {
            float xv = xb[t][g];
            aq += xv * Wq[g * 64 + h];
            ak += xv * Wk[g * 64 + h];
            av += xv * Wv[g * 64 + h];
        }
        qb[t][h] = aq; kb[t][h] = ak; vb[t][h] = av;
    }
    __syncthreads();
    {
        int t = h >> 3, s = h & 7;
        float a = 0.0f;
        for (int g = 0; g < 64; ++g) a += qb[t][g] * kb[s][g];
        aw[t][s] = a * 0.125f;
    }
    __syncthreads();
    if (h < SS) {
        float mx = aw[h][0];
        for (int s = 1; s < SS; ++s) mx = fmaxf(mx, aw[h][s]);
        float se = 0.0f;
        for (int s = 0; s < SS; ++s) { aw[h][s] = expf(aw[h][s] - mx); se += aw[h][s]; }
        float inv = 1.0f / se;
        for (int s = 0; s < SS; ++s) aw[h][s] *= inv;
    }
    __syncthreads();
    for (int t = 0; t < SS; ++t) {
        float a = 0.0f;
        for (int s = 0; s < SS; ++s) a += aw[t][s] * vb[s][h];
        ob[t][h] = a;
    }
    __syncthreads();
    for (int t = 0; t < SS; ++t) {
        float a = bo1[h];
        for (int g = 0; g < 64; ++g) a += ob[t][g] * Wo1[g * 64 + h];
        hb[t][h] = fmaxf(a, 0.0f);
    }
    __syncthreads();
    for (int t = 0; t < SS; ++t) {
        float a = bo2[h];
        for (int g = 0; g < 64; ++g) a += hb[t][g] * Wo2[g * 64 + h];
        xb[t][h] += a;
    }
    __syncthreads();
    if (h < SS) {
        float s = 0.0f;
        for (int g = 0; g < 64; ++g) s += xb[h][g];
        float m = s * (1.0f / 64.0f), v = 0.0f;
        for (int g = 0; g < 64; ++g) { float d = xb[h][g] - m; v += d * d; }
        mvar[h][0] = m;
        mvar[h][1] = 1.0f / sqrtf(v * (1.0f / 64.0f) + 1e-6f);
    }
    __syncthreads();
    float mx = -3.4e38f;
    for (int t = 0; t < SS; ++t) {
        float xv = (xb[t][h] - mvar[t][0]) * mvar[t][1] * lns[h] + lnb[h];
        mx = fmaxf(mx, xv);
    }
    xm[h] = mx;
    __syncthreads();
    if (h < LAT) {
        float a = blat[h];
        for (int g = 0; g < 64; ++g) a += xm[g] * Wlat[g * LAT + h];
        latb[h] = a;
    }
    __syncthreads();
    float a = bdin[h];
    for (int l = 0; l < LAT; ++l) a += latb[l] * Wdin[l * 64 + h];
    decx[(size_t)b * 64 + h] = fmaxf(a, 0.0f);
}

// ---------------------------------------------------------------------------
// Autoregressive decoder: 8 steps, one block of 64 threads per batch.
// ---------------------------------------------------------------------------
struct DecP {
    const float* qW[SS]; const float* qb[SS];
    const float* kW[SS]; const float* kb[SS];
    const float* vW[SS]; const float* vb[SS];
    const float* oW[SS]; const float* ob[SS];
};

__global__ void decoder_kernel(const float* __restrict__ decx, DecP dp,
                               float* __restrict__ decseq) {
    __shared__ float xd[SS + 1][64], kd[SS + 1][64], vd[SS + 1][64];
    __shared__ float qv[64], ov[64], aw[SS + 1];
    const int b = blockIdx.x;
    const int h = threadIdx.x;
    xd[0][h] = decx[(size_t)b * 64 + h];
    __syncthreads();

    for (int step = 0; step < SS; ++step) {
        const int L = step + 1;
        const float* Wq = dp.qW[step]; const float* bq = dp.qb[step];
        const float* Wk = dp.kW[step]; const float* bk = dp.kb[step];
        const float* Wv = dp.vW[step]; const float* bv = dp.vb[step];
        const float* Wo = dp.oW[step]; const float* bo = dp.ob[step];
        {
            float a = bq[h];
            for (int g = 0; g < 64; ++g) a += xd[L - 1][g] * Wq[g * 64 + h];
            qv[h] = a;
        }
        for (int s = 0; s < L; ++s) {
            float ak = bk[h], av = bv[h];
            for (int g = 0; g < 64; ++g) {
                float xv = xd[s][g];
                ak += xv * Wk[g * 64 + h];
                av += xv * Wv[g * 64 + h];
            }
            kd[s][h] = ak; vd[s][h] = av;
        }
        __syncthreads();
        if (h == 0) {
            float lg[SS + 1];
            float mxl = -3.4e38f;
            for (int s = 0; s < L; ++s) {
                float a = 0.0f;
                for (int g = 0; g < 64; ++g) a += qv[g] * kd[s][g];
                lg[s] = a * 0.125f;
                mxl = fmaxf(mxl, lg[s]);
            }
            float se = 0.0f;
            for (int s = 0; s < L; ++s) { lg[s] = expf(lg[s] - mxl); se += lg[s]; }
            float inv = 1.0f / se;
            for (int s = 0; s < L; ++s) aw[s] = lg[s] * inv;
        }
        __syncthreads();
        {
            float a = 0.0f;
            for (int s = 0; s < L; ++s) a += aw[s] * vd[s][h];
            ov[h] = fmaxf(a, 0.0f);
        }
        __syncthreads();
        {
            float a = bo[h];
            for (int g = 0; g < 64; ++g) a += ov[g] * Wo[g * 64 + h];
            xd[L][h] = xd[0][h] + a;
        }
        __syncthreads();
    }
    for (int t = 0; t < SS; ++t)
        decseq[((size_t)b * SS + t) * 64 + h] = xd[t + 1][h];
}

// dec_out: (B*S x 64) @ (64 x 3072) + b -> output
__global__ void decout_kernel(const float* __restrict__ seq, const float* __restrict__ W,
                              const float* __restrict__ b, float* __restrict__ out) {
    int i = blockIdx.x * blockDim.x + threadIdx.x;
    int j = i % 3072;
    int bs = i / 3072;
    const float* xr = seq + (size_t)bs * 64;
    float a = b[j];
    for (int g = 0; g < 64; ++g) a += xr[g] * W[(size_t)g * 3072 + j];
    out[i] = a;
}

// ---------------------------------------------------------------------------
extern "C" void kernel_launch(void* const* d_in, const int* in_sizes, int n_in,
                              void* d_out, int out_size, void* d_ws, size_t ws_size,
                              hipStream_t stream) {
    (void)in_sizes; (void)n_in; (void)out_size; (void)ws_size;
    auto F = [&](int i) { return (const float*)d_in[i]; };

    // input leaf indices (recursive dict-insertion-order flatten)
    const float* points = F(0);
    const float* encW = F(1); const float* encB = F(2);
    const int EL0 = 3, EL1 = 19, ETL = 35;
    const int LATI = 47, DIN = 49, DSTEP = 51, DOUT = 115;

    // workspace layout
    char* ws = (char*)d_ws;
    const size_t BUF = (size_t)P_TOTAL * 64 * sizeof(float);     // 8 MB
    float* x  = (float*)(ws + 0 * BUF);
    float* qb = (float*)(ws + 1 * BUF);
    float* kb = (float*)(ws + 2 * BUF);
    float* vb = (float*)(ws + 3 * BUF);
    float* pe = (float*)(ws + 4 * BUF);
    float* ao = (float*)(ws + 5 * BUF);
    int*   idx = (int*)(ws + 6 * BUF);
    char*  tail = ws + 6 * BUF + (size_t)P_TOTAL * KK * sizeof(int);
    float* xs     = (float*)tail;               tail += BB * SS * 64 * sizeof(float);
    float* decx   = (float*)tail;               tail += BB * 64 * sizeof(float);
    float* decseq = (float*)tail;

    const int GE = (P_TOTAL * 64) / 256;        // 8192
    const int GW = P_TOTAL / 16 / 8;            // 256
    const int GP = P_TOTAL / 8;                 // 4096

    dense3_kernel<<<GE, 256, 0, stream>>>(points, encW, encB, x);
    knn_kernel<<<BB * SS * 4, 256, 0, stream>>>(points, idx);

    for (int l = 0; l < 2; ++l) {
        const int b0 = (l == 0) ? EL0 : EL1;
        gemm64_qkv_wmma<<<GW, 256, 0, stream>>>(x, F(b0 + 0), F(b0 + 1), F(b0 + 2), F(b0 + 3),
                                                F(b0 + 4), F(b0 + 5), qb, kb, vb);
        dense3_kernel<<<GE, 256, 0, stream>>>(points, F(b0 + 6), F(b0 + 7), pe);
        ptattn_kernel<<<GP, 256, 0, stream>>>(qb, kb, vb, pe, points, idx,
                                              F(b0 + 8), F(b0 + 9), F(b0 + 10), F(b0 + 11), ao);
        gemm64_wmma<true, true><<<GW, 256, 0, stream>>>(ao, F(b0 + 12), F(b0 + 13), x, x);
        ln_kernel<<<GP, 256, 0, stream>>>(x, F(b0 + 14), F(b0 + 15));
    }

    maxpool_n_kernel<<<(BB * SS * 64) / 256, 256, 0, stream>>>(x, xs);

    etl_kernel<<<BB, 64, 0, stream>>>(xs,
        F(ETL + 0), F(ETL + 1), F(ETL + 2), F(ETL + 3), F(ETL + 4), F(ETL + 5),
        F(ETL + 6), F(ETL + 7), F(ETL + 8), F(ETL + 9), F(ETL + 10), F(ETL + 11),
        F(LATI), F(LATI + 1), F(DIN), F(DIN + 1), decx);

    DecP dp;
    for (int s = 0; s < SS; ++s) {
        const int bs = DSTEP + 8 * s;
        dp.qW[s] = F(bs + 0); dp.qb[s] = F(bs + 1);
        dp.kW[s] = F(bs + 2); dp.kb[s] = F(bs + 3);
        dp.vW[s] = F(bs + 4); dp.vb[s] = F(bs + 5);
        dp.oW[s] = F(bs + 6); dp.ob[s] = F(bs + 7);
    }
    decoder_kernel<<<BB, 64, 0, stream>>>(decx, dp, decseq);

    decout_kernel<<<(BB * SS * 3072) / 256, 256, 0, stream>>>(decseq, F(DOUT), F(DOUT + 1),
                                                              (float*)d_out);
}
